// Model_xLSTM_39221641347095
// MI455X (gfx1250) — matrix operand order repaired and checked by reading, not compiled, not run on hardware
//
#include <hip/hip_runtime.h>
#include <hip/hip_bf16.h>
#include <cstdint>
#include <cstddef>

// ---------------- problem constants ----------------
constexpr int cB  = 128;
constexpr int cS  = 256;
constexpr int cNF = 32;
constexpr int cE  = 4;
constexpr int cD  = 256;
constexpr int cH  = 4;
constexpr int cDI = 512;
constexpr int cDHM = 128;
constexpr int cDHS = 64;
constexpr int cFF = 352;
constexpr int cBS = cB * cS;

// ---------------- CDNA5 WMMA types ----------------
typedef __bf16 bf16_t;
typedef __attribute__((ext_vector_type(16))) __bf16 v16bf;
typedef __attribute__((ext_vector_type(8)))  float  v8f;

static __device__ __forceinline__ bf16_t bfc(float f) { return (bf16_t)f; }

static __device__ __forceinline__ v8f wmma_bf16(v16bf a, v16bf b, v8f c) {
  return __builtin_amdgcn_wmma_f32_16x16x32_bf16(false, a, false, b, (short)0, c, false, false);
}

static __device__ __forceinline__ float sigmoidf_(float x) { return 1.f / (1.f + __expf(-x)); }
static __device__ __forceinline__ float siluf_(float x)    { return x * sigmoidf_(x); }

// ---------------- CDNA5 async global->LDS copy (ASYNCcnt) ----------------
// LDS offset = low 32 bits of the generic shared-aperture address.
static __device__ __forceinline__ uint32_t lds_off32(const void* p) {
  return (uint32_t)(size_t)p;
}
// per-lane 16-byte async copy: LDS[dst] = MEM[src]
static __device__ __forceinline__ void async_copy_b128(uint32_t lds_dst, const void* gsrc) {
  asm volatile("global_load_async_to_lds_b128 %0, %1, off"
               :: "v"(lds_dst), "v"(gsrc) : "memory");
}
static __device__ __forceinline__ void wait_async0() {
  asm volatile("s_wait_asynccnt 0x0" ::: "memory");
}

// =====================================================================
// 1. embedding gather + input projection: x = [x_sens|e_esn|e_prof] @ W_in + b
// =====================================================================
__global__ __launch_bounds__(256) void embed_kernel(
    const float* __restrict__ xs, const int* __restrict__ esn, const int* __restrict__ prof,
    const float* __restrict__ Eesn, const float* __restrict__ Eprof,
    const float* __restrict__ Win, const float* __restrict__ bin,
    float* __restrict__ x)
{
  const int bs = blockIdx.x;
  const int b = bs / cS;
  const int tid = threadIdx.x;
  __shared__ float inr[40];
  if (tid < cNF)                inr[tid] = xs[(size_t)bs * cNF + tid];
  else if (tid < cNF + cE)      inr[tid] = Eesn[(size_t)esn[b] * cE + (tid - cNF)];
  else if (tid < cNF + 2 * cE)  inr[tid] = Eprof[(size_t)prof[b] * cE + (tid - cNF - cE)];
  __syncthreads();
  float acc = bin[tid];
  #pragma unroll
  for (int j = 0; j < cNF + 2 * cE; ++j) acc += inr[j] * Win[j * cD + tid];
  x[(size_t)bs * cD + tid] = acc;
}

// =====================================================================
// 2. LayerNorm over D=256 (one row per block)
// =====================================================================
__global__ __launch_bounds__(256) void ln_kernel(
    const float* __restrict__ in, const float* __restrict__ g, const float* __restrict__ bb,
    float* __restrict__ out, float eps)
{
  const int row = blockIdx.x;
  const int tid = threadIdx.x;
  float v = in[(size_t)row * cD + tid];
  float s1 = v, s2 = v * v;
  #pragma unroll
  for (int m = 1; m < 32; m <<= 1) { s1 += __shfl_xor(s1, m); s2 += __shfl_xor(s2, m); }
  __shared__ float as1[8], as2[8];
  if ((tid & 31) == 0) { as1[tid >> 5] = s1; as2[tid >> 5] = s2; }
  __syncthreads();
  float S1 = 0.f, S2 = 0.f;
  #pragma unroll
  for (int i = 0; i < 8; ++i) { S1 += as1[i]; S2 += as2[i]; }
  const float mu = S1 * (1.f / cD);
  const float var = S2 * (1.f / cD) - mu * mu;
  out[(size_t)row * cD + tid] = (v - mu) * rsqrtf(var + eps) * g[tid] + bb[tid];
}

// =====================================================================
// 3. bf16-WMMA GEMM: C = A(MxK) @ W(KxN) + bias [+ resid]
//    128x128 block tile, 8 waves, wave = 32x64 -> 8 WMMA / k-step.
//    OUTB16: store bf16 (for q/k/v feeding attention).
//    Preconditions: M % 128 == 0, K % 32 == 0, lda/ldw multiples of 4;
//    only N may be ragged (352).
// =====================================================================
template<bool RESID, bool OUTB16>
__global__ __launch_bounds__(256) void gemm_kernel(
    const float* __restrict__ A, int lda,
    const float* __restrict__ W, int ldw,
    const float* __restrict__ bias,
    const float* __restrict__ resid, int ldr,
    void* __restrict__ Cout, int ldc,
    int M, int N, int Kt)
{
  __shared__ bf16_t As[128][34];   // [m][k]
  __shared__ bf16_t Ws[128][34];   // [n][k] (transposed on load)
  const int tid = threadIdx.x;
  const int wave = tid >> 5, lane = tid & 31;
  const int colL = lane & 15, hl = lane >> 4;
  const int m0 = blockIdx.y * 128;
  const int n0 = blockIdx.x * 128;
  const int wm = (wave & 3) * 32;          // wave row block (2 x 16)
  const int wn = (wave >> 2) * 64;         // wave col block (4 x 16)
  const v8f vzero = {0.f, 0.f, 0.f, 0.f, 0.f, 0.f, 0.f, 0.f};
  v8f acc[2][4];
  #pragma unroll
  for (int i = 0; i < 2; ++i)
    #pragma unroll
    for (int j = 0; j < 4; ++j) acc[i][j] = vzero;

  for (int k0 = 0; k0 < Kt; k0 += 32) {
    // --- A tile 128x32, unguarded float4 loads (global_load_b128) ---
    #pragma unroll
    for (int v = 0; v < 4; ++v) {
      const int idx = tid + v * 256;          // 1024 float4s
      const int r = idx >> 3, c4 = (idx & 7) * 4;
      const float4 val = *(const float4*)&A[(size_t)(m0 + r) * lda + k0 + c4];
      As[r][c4 + 0] = bfc(val.x);
      As[r][c4 + 1] = bfc(val.y);
      As[r][c4 + 2] = bfc(val.z);
      As[r][c4 + 3] = bfc(val.w);
    }
    // --- W tile 32x128 -> transposed [n][k] ---
    #pragma unroll
    for (int v = 0; v < 4; ++v) {
      const int idx = tid + v * 256;          // 1024 float4s
      const int r = idx >> 5, c4 = (idx & 31) * 4;
      const int gk = k0 + r;
      const int gn = n0 + c4;
      if (gn + 3 < N) {
        const float4 val = *(const float4*)&W[(size_t)gk * ldw + gn];
        Ws[c4 + 0][r] = bfc(val.x);
        Ws[c4 + 1][r] = bfc(val.y);
        Ws[c4 + 2][r] = bfc(val.z);
        Ws[c4 + 3][r] = bfc(val.w);
      } else {
        #pragma unroll
        for (int q = 0; q < 4; ++q)
          Ws[c4 + q][r] = bfc((gn + q < N) ? W[(size_t)gk * ldw + gn + q] : 0.f);
      }
    }
    if (k0 + 32 < Kt) {  // speculative pull of next tiles (global_prefetch)
      __builtin_prefetch(&A[(size_t)(m0 + (tid >> 1)) * lda + k0 + 32], 0, 1);
      __builtin_prefetch(&W[(size_t)(k0 + 32 + (tid >> 3)) * ldw + n0], 0, 1);
    }
    __syncthreads();
    // A fragments (16x32): lane row = lane&15; k halves by lane>>4
    const int kh = hl * 8;
    v16bf af[2];
    #pragma unroll
    for (int i = 0; i < 2; ++i) {
      const int arow = wm + i * 16 + colL;
      #pragma unroll
      for (int e = 0; e < 8; ++e) {
        af[i][e]     = As[arow][2 * kh + e];
        af[i][e + 8] = As[arow][2 * kh + 8 + e];
      }
    }
    // B fragments (32x16): lane col = lane&15; k = (lane>>4)*16 + e
    const int kb = hl * 16;
    v16bf bfr[4];
    #pragma unroll
    for (int j = 0; j < 4; ++j) {
      const int brow = wn + j * 16 + colL;
      #pragma unroll
      for (int e = 0; e < 16; ++e) bfr[j][e] = Ws[brow][kb + e];
    }
    #pragma unroll
    for (int i = 0; i < 2; ++i)
      #pragma unroll
      for (int j = 0; j < 4; ++j)
        acc[i][j] = wmma_bf16(af[i], bfr[j], acc[i][j]);
    __syncthreads();
  }
  // epilogue: value (m,n): vgpr g = m&7, lane = n + 16*(m>>3)
  #pragma unroll
  for (int i = 0; i < 2; ++i) {
    #pragma unroll
    for (int j = 0; j < 4; ++j) {
      const int gn = n0 + wn + j * 16 + colL;
      if (gn >= N) continue;
      const float bv = bias ? bias[gn] : 0.f;
      #pragma unroll
      for (int g = 0; g < 8; ++g) {
        const int gm = m0 + wm + i * 16 + 8 * hl + g;
        float val = acc[i][j][g] + bv;
        if (RESID) val += resid[(size_t)gm * ldr + gn];
        if (OUTB16) ((bf16_t*)Cout)[(size_t)gm * ldc + gn] = bfc(val);
        else        ((float*)Cout)[(size_t)gm * ldc + gn] = val;
      }
    }
  }
}

// =====================================================================
// 4. Causal depthwise conv (K=4) + SiLU
// =====================================================================
__global__ __launch_bounds__(256) void conv_silu_kernel(
    const float* __restrict__ in, const float* __restrict__ w,
    const float* __restrict__ bias, float* __restrict__ out, int C)
{
  const int bs = blockIdx.x;
  const int b = bs / cS, s = bs % cS;
  for (int c = threadIdx.x; c < C; c += blockDim.x) {
    float acc = bias[c];
    #pragma unroll
    for (int j = 0; j < 4; ++j) {
      const int sp = s - 3 + j;
      const float xv = (sp >= 0) ? in[((size_t)(b * cS + sp)) * C + c] : 0.f;
      acc += xv * w[j * C + c];
    }
    out[(size_t)bs * C + c] = siluf_(acc);
  }
}

// =====================================================================
// 5. i/f gate pre-activations: [q|k|v](1536) @ W(1536x4) + b  (per b,s)
//    q/k/v are bf16.
// =====================================================================
__global__ __launch_bounds__(256) void gates_kernel(
    const bf16_t* __restrict__ q, const bf16_t* __restrict__ k, const bf16_t* __restrict__ v,
    const float* __restrict__ Wi, const float* __restrict__ bi,
    const float* __restrict__ Wf, const float* __restrict__ bfv,
    float* __restrict__ ipre, float* __restrict__ fpre)
{
  const int bs = blockIdx.x;
  const int tid = threadIdx.x;
  float ai[4] = {0.f, 0.f, 0.f, 0.f}, af[4] = {0.f, 0.f, 0.f, 0.f};
  const bf16_t* qr = q + (size_t)bs * cDI;
  const bf16_t* kr = k + (size_t)bs * cDI;
  const bf16_t* vr = v + (size_t)bs * cDI;
  for (int j = tid; j < cDI; j += 256) {
    const float qv = (float)qr[j], kv = (float)kr[j], vv = (float)vr[j];
    #pragma unroll
    for (int h = 0; h < 4; ++h) {
      ai[h] += qv * Wi[(size_t)j * 4 + h] + kv * Wi[(size_t)(cDI + j) * 4 + h] + vv * Wi[(size_t)(2 * cDI + j) * 4 + h];
      af[h] += qv * Wf[(size_t)j * 4 + h] + kv * Wf[(size_t)(cDI + j) * 4 + h] + vv * Wf[(size_t)(2 * cDI + j) * 4 + h];
    }
  }
  __shared__ float red[8][8];
  #pragma unroll
  for (int t = 0; t < 8; ++t) {
    float val = (t < 4) ? ai[t] : af[t - 4];
    #pragma unroll
    for (int m = 1; m < 32; m <<= 1) val += __shfl_xor(val, m);
    if ((tid & 31) == 0) red[tid >> 5][t] = val;
  }
  __syncthreads();
  if (tid < 8) {
    float tot = 0.f;
    #pragma unroll
    for (int w = 0; w < 8; ++w) tot += red[w][tid];
    if (tid < 4) ipre[(size_t)bs * cH + tid] = tot + bi[tid];
    else         fpre[(size_t)bs * cH + (tid - 4)] = tot + bfv[tid - 4];
  }
}

// =====================================================================
// 6. Per-(b,h) scans: F = cumsum(log_sigmoid(fpre)),
//    mstab = F + cummax(ipre - F); also rearranged ipre
// =====================================================================
__global__ __launch_bounds__(256) void scan_kernel(
    const float* __restrict__ fpre, const float* __restrict__ ipre,
    float* __restrict__ FT, float* __restrict__ MS, float* __restrict__ IPT)
{
  const int bh = blockIdx.x;
  const int b = bh >> 2, h = bh & 3;
  const int s = threadIdx.x;
  __shared__ float buf[256];
  const float f = fpre[((size_t)(b * cS + s)) * cH + h];
  const float ip = ipre[((size_t)(b * cS + s)) * cH + h];
  const float lfg = (f >= 0.f) ? -log1pf(__expf(-f)) : (f - log1pf(__expf(f)));
  buf[s] = lfg;
  __syncthreads();
  for (int off = 1; off < 256; off <<= 1) {
    const float add = (s >= off) ? buf[s - off] : 0.f;
    __syncthreads();
    buf[s] += add;
    __syncthreads();
  }
  const float F = buf[s];
  FT[(size_t)bh * cS + s] = F;
  IPT[(size_t)bh * cS + s] = ip;
  buf[s] = ip - F;
  __syncthreads();
  for (int off = 1; off < 256; off <<= 1) {
    const float mx = (s >= off) ? buf[s - off] : -3.4e38f;
    __syncthreads();
    buf[s] = fmaxf(buf[s], mx);
    __syncthreads();
  }
  MS[(size_t)bh * cS + s] = F + buf[s];
}

// =====================================================================
// 7. mLSTM attention, fully fused per (b,h); q/k/v already bf16.
//    K chunk staged into LDS via async global->LDS b128 copies (ASYNCcnt),
//    V chunk transposed manually; scores & C@V via WMMA; fused decay mask,
//    nrm, headnorm, +skip*a, *silu(zg) -> t1.
//    512 threads = 16 waves, wave w owns t rows [16w, 16w+16)
// =====================================================================
__global__ __launch_bounds__(512) void attn_kernel(
    const bf16_t* __restrict__ q, const bf16_t* __restrict__ k, const bf16_t* __restrict__ v,
    const float* __restrict__ FT, const float* __restrict__ MS, const float* __restrict__ IPT,
    const float* __restrict__ ng, const float* __restrict__ nb, const float* __restrict__ skip,
    const float* __restrict__ aact, const float* __restrict__ zg, float* __restrict__ t1)
{
  const int b = blockIdx.x, h = blockIdx.y;
  const int tid = threadIdx.x;
  const int wave = tid >> 5, lane = tid & 31;
  const int colL = lane & 15, hl = lane >> 4;
  const int t0 = wave * 16;
  const int kh = hl * 8;

  __shared__ bf16_t Ks[64][136];    // K chunk, [s][d], padded rows (17x16B)
  __shared__ bf16_t Vt[128][72];    // V chunk transposed, [d][s]
  __shared__ bf16_t Cs[16][512];    // per-wave 16x32 score relayout bounce
  __shared__ float Fs[256], Ms[256], Ip[256];

  const size_t base = ((size_t)(b * cH + h)) * cS;
  if (tid < 256) { Fs[tid] = FT[base + tid]; Ms[tid] = MS[base + tid]; Ip[tid] = IPT[base + tid]; }

  // q fragments (4 k-steps of 32 over DHM=128), held for whole kernel
  v16bf qf[4];
  {
    const int tq = t0 + colL;
    const bf16_t* qr = q + ((size_t)(b * cS + tq)) * cDI + h * cDHM;
    #pragma unroll
    for (int kc = 0; kc < 4; ++kc)
      #pragma unroll
      for (int e = 0; e < 8; ++e) {
        qf[kc][e]     = qr[kc * 32 + 2 * kh + e];
        qf[kc][e + 8] = qr[kc * 32 + 2 * kh + 8 + e];
      }
  }

  const v8f vzero = {0.f, 0.f, 0.f, 0.f, 0.f, 0.f, 0.f, 0.f};
  v8f outv[8];
  #pragma unroll
  for (int i = 0; i < 8; ++i) outv[i] = vzero;
  float rsum[8] = {0.f, 0.f, 0.f, 0.f, 0.f, 0.f, 0.f, 0.f};
  const float invs = 0.088388347648318447f;  // 1/sqrt(128)

  for (int scn = 0; scn < 4; ++scn) {
    __syncthreads();
    // ---- K chunk: 64 rows x 256B via async global->LDS (padded dest) ----
    #pragma unroll
    for (int it2 = 0; it2 < 2; ++it2) {
      const int idx = tid + it2 * 512;        // 1024 chunks of 16B
      const int r = idx >> 4, j = idx & 15;   // row, 16B-chunk within row
      const bf16_t* gsrc = k + ((size_t)(b * cS + scn * 64 + r)) * cDI + h * cDHM + j * 8;
      async_copy_b128(lds_off32(&Ks[r][j * 8]), gsrc);
    }
    // ---- V chunk: manual bf16 transpose into LDS ----
    for (int i = tid; i < 64 * 64; i += 512) {   // uint32 = 2 bf16 elems
      const int r = i >> 6, c = (i & 63) * 2;
      const uint32_t u = *(const uint32_t*)&v[((size_t)(b * cS + scn * 64 + r)) * cDI + h * cDHM + c];
      Vt[c][r]     = ((const bf16_t*)&u)[0];
      Vt[c + 1][r] = ((const bf16_t*)&u)[1];
    }
    wait_async0();
    __syncthreads();
    for (int sb = 0; sb < 2; ++sb) {
      const int s0 = scn * 64 + sb * 32;
      if (s0 > t0 + 15) break;     // wave-uniform causal skip (EXEC stays full)
      const int sl = sb * 32;
      v8f p0 = vzero, p1 = vzero;
      #pragma unroll
      for (int kc = 0; kc < 4; ++kc) {
        v16bf b0, b1;
        #pragma unroll
        for (int e = 0; e < 16; ++e) {
          b0[e] = Ks[sl + colL][kc * 32 + hl * 16 + e];
          b1[e] = Ks[sl + 16 + colL][kc * 32 + hl * 16 + e];
        }
        p0 = wmma_bf16(qf[kc], b0, p0);
        p1 = wmma_bf16(qf[kc], b1, p1);
      }
      // decay mask + stabilizer, row sums, relayout to A-frag via LDS
      #pragma unroll
      for (int g = 0; g < 8; ++g) {
        const int m = g + 8 * hl;
        const int tt = t0 + m;
        const float Ft = Fs[tt], Mt = Ms[tt];
        int ss = s0 + colL;
        float d0 = (ss <= tt) ? __expf(Ft - Fs[ss] + Ip[ss] - Mt) : 0.f;
        const float c0 = p0[g] * invs * d0;
        rsum[g] += c0;
        Cs[wave][m * 32 + colL] = bfc(c0);
        ss = s0 + 16 + colL;
        float d1 = (ss <= tt) ? __expf(Ft - Fs[ss] + Ip[ss] - Mt) : 0.f;
        const float c1 = p1[g] * invs * d1;
        rsum[g] += c1;
        Cs[wave][m * 32 + 16 + colL] = bfc(c1);
      }
      v16bf caf;
      #pragma unroll
      for (int e = 0; e < 8; ++e) {
        caf[e]     = Cs[wave][colL * 32 + 2 * kh + e];
        caf[e + 8] = Cs[wave][colL * 32 + 2 * kh + 8 + e];
      }
      // C(16x32) @ V(32x128): 8 output n-tiles
      #pragma unroll
      for (int nt = 0; nt < 8; ++nt) {
        v16bf vfrag;
        #pragma unroll
        for (int e = 0; e < 16; ++e)
          vfrag[e] = Vt[nt * 16 + colL][sl + hl * 16 + e];
        outv[nt] = wmma_bf16(caf, vfrag, outv[nt]);
      }
    }
  }

  // nrm = max(|sum_s C|, exp(-mstab))
  float invn[8];
  #pragma unroll
  for (int g = 0; g < 8; ++g) {
    float r = rsum[g];
    r += __shfl_xor(r, 1); r += __shfl_xor(r, 2);
    r += __shfl_xor(r, 4); r += __shfl_xor(r, 8);
    const int tt = t0 + 8 * hl + g;
    invn[g] = 1.f / fmaxf(fabsf(r), __expf(-Ms[tt]));
  }
  // headnorm stats per row over DHM=128
  float mu[8], var[8];
  #pragma unroll
  for (int g = 0; g < 8; ++g) {
    float sm = 0.f, sq = 0.f;
    #pragma unroll
    for (int nt = 0; nt < 8; ++nt) {
      const float val = outv[nt][g] * invn[g];
      sm += val; sq += val * val;
    }
    sm += __shfl_xor(sm, 1); sm += __shfl_xor(sm, 2); sm += __shfl_xor(sm, 4); sm += __shfl_xor(sm, 8);
    sq += __shfl_xor(sq, 1); sq += __shfl_xor(sq, 2); sq += __shfl_xor(sq, 4); sq += __shfl_xor(sq, 8);
    mu[g] = sm * (1.f / 128.f);
    var[g] = sq * (1.f / 128.f) - mu[g] * mu[g];
  }
  // fused epilogue: headnorm + skip*a, * silu(zg) -> t1
  #pragma unroll
  for (int nt = 0; nt < 8; ++nt) {
    const int d = nt * 16 + colL;
    const int hd = h * cDHM + d;
    const float gw = ng[hd], gb = nb[hd], sk = skip[hd];
    #pragma unroll
    for (int g = 0; g < 8; ++g) {
      const int tt = t0 + 8 * hl + g;
      const size_t gi = ((size_t)(b * cS + tt)) * cDI + hd;
      const float val = outv[nt][g] * invn[g];
      float hn = (val - mu[g]) * rsqrtf(var[g] + 1e-6f) * gw + gb;
      hn += sk * aact[gi];
      t1[gi] = hn * siluf_(zg[gi]);
    }
  }
}

// =====================================================================
// 8. sLSTM sequential scan: one block per batch row, thread = (h,e).
//    All four recurrent matrices (256 KB) pulled into dynamic LDS with
//    async global->LDS copies — CDNA5 WGP has 320 KB LDS.
// =====================================================================
__global__ __launch_bounds__(256) void slstm_kernel(
    const float* __restrict__ it, const float* __restrict__ ft,
    const float* __restrict__ zt, const float* __restrict__ ot,
    const float* __restrict__ Ri, const float* __restrict__ Rf,
    const float* __restrict__ Rz, const float* __restrict__ Ro,
    float* __restrict__ hs)
{
  extern __shared__ float Rsh[];           // 4 gates x (H*64*64) floats
  const int b = blockIdx.x;
  const int tid = threadIdx.x;
  const int h = tid >> 6, e = tid & 63;
  __shared__ float hsh[4][64];

  const float* __restrict__ Rsrc[4] = {Ri, Rf, Rz, Ro};
  // 256 KB = 16384 x 16B chunks, async global->LDS
  for (int i = tid; i < 16384; i += 256) {
    const int gix = i >> 12;                  // 4096 chunks per gate
    const int rem = i & 4095;
    async_copy_b128(lds_off32(&Rsh[(size_t)gix * (cH * cDHS * cDHS) + (size_t)rem * 4]),
                    Rsrc[gix] + (size_t)rem * 4);
  }

  float c = 0.f, n = 0.f, mstate = 0.f;
  hsh[h][e] = 0.f;
  const int rbase = h * cDHS * cDHS + e;
  const float* __restrict__ Li = Rsh + 0 * (cH * cDHS * cDHS) + rbase;
  const float* __restrict__ Lf = Rsh + 1 * (cH * cDHS * cDHS) + rbase;
  const float* __restrict__ Lz = Rsh + 2 * (cH * cDHS * cDHS) + rbase;
  const float* __restrict__ Lo = Rsh + 3 * (cH * cDHS * cDHS) + rbase;
  wait_async0();
  __syncthreads();

  for (int s = 0; s < cS; ++s) {
    float ri = 0.f, rf = 0.f, rz = 0.f, ro = 0.f;
    #pragma unroll 8
    for (int d = 0; d < 64; ++d) {
      const float hp = hsh[h][d];
      ri += hp * Li[d * 64];
      rf += hp * Lf[d * 64];
      rz += hp * Lz[d * 64];
      ro += hp * Lo[d * 64];
    }
    const size_t gi = ((size_t)(b * cS + s)) * cD + h * 64 + e;
    const float ip = it[gi] + ri;
    const float fp = ft[gi] + rf;
    const float zp = tanhf(zt[gi] + rz);
    const float op = sigmoidf_(ot[gi] + ro);
    const float mn = fmaxf(fp + mstate, ip);
    const float ig = __expf(ip - mn);
    const float fg = __expf(fp + mstate - mn);
    c = fg * c + ig * zp;
    n = fg * n + ig;
    mstate = mn;
    const float hv = op * c / n;
    hs[gi] = hv;
    __syncthreads();
    hsh[h][e] = hv;
    __syncthreads();
  }
}

// =====================================================================
// 9. headnorm(hs) over DHS=64 + residual add into x
// =====================================================================
__global__ __launch_bounds__(64) void hnorm_add_kernel(
    const float* __restrict__ hs, const float* __restrict__ g, const float* __restrict__ bb,
    float* __restrict__ x)
{
  const int row = blockIdx.x;           // bs*H + h
  const int bs = row >> 2, h = row & 3;
  const int e = threadIdx.x;
  const float v = hs[(size_t)bs * cD + h * cDHS + e];
  float s1 = v, s2 = v * v;
  #pragma unroll
  for (int m = 1; m < 32; m <<= 1) { s1 += __shfl_xor(s1, m); s2 += __shfl_xor(s2, m); }
  __shared__ float rs[2], rq[2];
  if ((e & 31) == 0) { rs[e >> 5] = s1; rq[e >> 5] = s2; }
  __syncthreads();
  const float S1 = rs[0] + rs[1], S2 = rq[0] + rq[1];
  const float mu = S1 * (1.f / 64.f), var = S2 * (1.f / 64.f) - mu * mu;
  x[(size_t)bs * cD + h * cDHS + e] +=
      (v - mu) * rsqrtf(var + 1e-6f) * g[h * cDHS + e] + bb[h * cDHS + e];
}

// =====================================================================
// 10. GLU: t = gelu_tanh(g) * u
// =====================================================================
__global__ void gelu_mul_kernel(const float* __restrict__ g_, const float* __restrict__ u_,
                                float* __restrict__ o_, int n)
{
  const int i = blockIdx.x * blockDim.x + threadIdx.x;
  if (i < n) {
    const float x = g_[i];
    const float t = tanhf(0.7978845608028654f * (x + 0.044715f * x * x * x));
    o_[i] = 0.5f * x * (1.f + t) * u_[i];
  }
}

// =====================================================================
// 11. head MLP on last token
// =====================================================================
__global__ __launch_bounds__(64) void head_kernel(
    const float* __restrict__ x, const float* __restrict__ W1, const float* __restrict__ b1,
    const float* __restrict__ W2, const float* __restrict__ b2, float* __restrict__ out)
{
  const int b = blockIdx.x;
  const int j = threadIdx.x;
  const float* last = x + ((size_t)(b * cS + (cS - 1))) * cD;
  float acc = b1[j];
  for (int d = 0; d < cD; ++d) acc += last[d] * W1[d * 64 + j];
  acc = fmaxf(acc, 0.f);
  __shared__ float hid[64];
  hid[j] = acc;
  __syncthreads();
  if (j < 3) {
    float o = b2[j];
    #pragma unroll
    for (int jj = 0; jj < 64; ++jj) o += hid[jj] * W2[jj * 3 + j];
    out[b * 3 + j] = o;
  }
}

// =====================================================================
// host orchestration
// =====================================================================
extern "C" void kernel_launch(void* const* d_in, const int* in_sizes, int n_in,
                              void* d_out, int out_size, void* d_ws, size_t ws_size,
                              hipStream_t stream) {
  (void)in_sizes; (void)n_in; (void)out_size; (void)ws_size;
  const float* x_sens  = (const float*)d_in[0];
  const int*   x_esn   = (const int*)  d_in[1];
  const int*   x_prof  = (const int*)  d_in[2];
  const float* emb_esn = (const float*)d_in[3];
  const float* emb_prof= (const float*)d_in[4];
  const float* W_in    = (const float*)d_in[5];
  const float* b_in    = (const float*)d_in[6];
  const float* m_ln_g  = (const float*)d_in[7];
  const float* m_ln_b  = (const float*)d_in[8];
  const float* m_Wup   = (const float*)d_in[9];
  const float* m_bup   = (const float*)d_in[10];
  const float* m_convW = (const float*)d_in[11];
  const float* m_convb = (const float*)d_in[12];
  const float* m_Wq    = (const float*)d_in[13];
  const float* m_bq    = (const float*)d_in[14];
  const float* m_Wk    = (const float*)d_in[15];
  const float* m_bk    = (const float*)d_in[16];
  const float* m_Wv    = (const float*)d_in[17];
  const float* m_bv    = (const float*)d_in[18];
  const float* m_Wi    = (const float*)d_in[19];
  const float* m_bi    = (const float*)d_in[20];
  const float* m_Wf    = (const float*)d_in[21];
  const float* m_bf    = (const float*)d_in[22];
  const float* m_skip  = (const float*)d_in[23];
  const float* m_ng    = (const float*)d_in[24];
  const float* m_nb    = (const float*)d_in[25];
  const float* m_Wdown = (const float*)d_in[26];
  const float* m_bdown = (const float*)d_in[27];
  const float* s_ln_g  = (const float*)d_in[28];
  const float* s_ln_b  = (const float*)d_in[29];
  const float* s_convW = (const float*)d_in[30];
  const float* s_convb = (const float*)d_in[31];
  const float* s_Wi    = (const float*)d_in[32];
  const float* s_bi    = (const float*)d_in[33];
  const float* s_Wf    = (const float*)d_in[34];
  const float* s_bf    = (const float*)d_in[35];
  const float* s_Wz    = (const float*)d_in[36];
  const float* s_bz    = (const float*)d_in[37];
  const float* s_Wo    = (const float*)d_in[38];
  const float* s_bo    = (const float*)d_in[39];
  const float* s_Ri    = (const float*)d_in[40];
  const float* s_Rf    = (const float*)d_in[41];
  const float* s_Rz    = (const float*)d_in[42];
  const float* s_Ro    = (const float*)d_in[43];
  const float* s_ng    = (const float*)d_in[44];
  const float* s_nb    = (const float*)d_in[45];
  const float* f_ln_g  = (const float*)d_in[46];
  const float* f_ln_b  = (const float*)d_in[47];
  const float* f_Wup   = (const float*)d_in[48];
  const float* f_bup   = (const float*)d_in[49];
  const float* f_Wdown = (const float*)d_in[50];
  const float* f_bdown = (const float*)d_in[51];
  const float* h_W1    = (const float*)d_in[52];
  const float* h_b1    = (const float*)d_in[53];
  const float* h_W2    = (const float*)d_in[54];
  const float* h_b2    = (const float*)d_in[55];

  float* ws = (float*)d_ws;
  size_t off = 0;
  auto alloc = [&](size_t n) { float* p = ws + off; off += n; return p; };
  float* x    = alloc((size_t)cBS * cD);
  float* xn   = alloc((size_t)cBS * cD);
  float* xm   = alloc((size_t)cBS * cDI);   // reused: it (D)
  float* zgb  = alloc((size_t)cBS * cDI);   // reused: ft (D)
  float* ab   = alloc((size_t)cBS * cDI);   // a / ac / u_
  float* qreg = alloc((size_t)cBS * cD);    // bf16 q   -> later f32 hs
  float* kreg = alloc((size_t)cBS * cD);    // bf16 k   -> later f32 zt
  float* vreg = alloc((size_t)cBS * cD);    // bf16 v   -> later f32 ot
  float* t1   = alloc((size_t)cBS * cDI);   // mLSTM gated out / g_ / t2
  float* ipre = alloc((size_t)cBS * cH);
  float* fpre = alloc((size_t)cBS * cH);
  float* FT   = alloc((size_t)cB * cH * cS);
  float* MSb  = alloc((size_t)cB * cH * cS);
  float* IPT  = alloc((size_t)cB * cH * cS);
  bf16_t* qb16 = (bf16_t*)qreg;
  bf16_t* kb16 = (bf16_t*)kreg;
  bf16_t* vb16 = (bf16_t*)vreg;

  const dim3 gN512(4, cBS / 128), gN256(2, cBS / 128), gN352(3, cBS / 128);
  const size_t slstmLds = (size_t)4 * cH * cDHS * cDHS * sizeof(float);  // 256 KB

  // ---- input stage ----
  embed_kernel<<<cBS, 256, 0, stream>>>(x_sens, x_esn, x_prof, emb_esn, emb_prof, W_in, b_in, x);

  // ---- mLSTM block ----
  ln_kernel<<<cBS, 256, 0, stream>>>(x, m_ln_g, m_ln_b, xn, 1e-5f);
  gemm_kernel<false, false><<<gN512, 256, 0, stream>>>(xn, cD, m_Wup,       2 * cDI, m_bup,       nullptr, 0, xm,  cDI, cBS, cDI, cD);
  gemm_kernel<false, false><<<gN512, 256, 0, stream>>>(xn, cD, m_Wup + cDI, 2 * cDI, m_bup + cDI, nullptr, 0, zgb, cDI, cBS, cDI, cD);
  conv_silu_kernel<<<cBS, 256, 0, stream>>>(xm, m_convW, m_convb, ab, cDI);
  gemm_kernel<false, true><<<gN512, 256, 0, stream>>>(ab, cDI, m_Wq, cDI, m_bq, nullptr, 0, qb16, cDI, cBS, cDI, cDI);
  gemm_kernel<false, true><<<gN512, 256, 0, stream>>>(ab, cDI, m_Wk, cDI, m_bk, nullptr, 0, kb16, cDI, cBS, cDI, cDI);
  gemm_kernel<false, true><<<gN512, 256, 0, stream>>>(xm, cDI, m_Wv, cDI, m_bv, nullptr, 0, vb16, cDI, cBS, cDI, cDI);
  gates_kernel<<<cBS, 256, 0, stream>>>(qb16, kb16, vb16, m_Wi, m_bi, m_Wf, m_bf, ipre, fpre);
  scan_kernel<<<cB * cH, 256, 0, stream>>>(fpre, ipre, FT, MSb, IPT);
  attn_kernel<<<dim3(cB, cH), 512, 0, stream>>>(qb16, kb16, vb16, FT, MSb, IPT,
                                                m_ng, m_nb, m_skip, ab, zgb, t1);
  gemm_kernel<true, false><<<gN256, 256, 0, stream>>>(t1, cDI, m_Wdown, cD, m_bdown, x, cD, x, cD, cBS, cD, cDI);

  // ---- sLSTM block ----
  ln_kernel<<<cBS, 256, 0, stream>>>(x, s_ln_g, s_ln_b, xn, 1e-5f);
  conv_silu_kernel<<<cBS, 256, 0, stream>>>(xn, s_convW, s_convb, ab, cD);
  gemm_kernel<false, false><<<gN256, 256, 0, stream>>>(ab, cD, s_Wi, cD, s_bi, nullptr, 0, xm,   cD, cBS, cD, cD);
  gemm_kernel<false, false><<<gN256, 256, 0, stream>>>(ab, cD, s_Wf, cD, s_bf, nullptr, 0, zgb,  cD, cBS, cD, cD);
  gemm_kernel<false, false><<<gN256, 256, 0, stream>>>(xn, cD, s_Wz, cD, s_bz, nullptr, 0, kreg, cD, cBS, cD, cD);
  gemm_kernel<false, false><<<gN256, 256, 0, stream>>>(xn, cD, s_Wo, cD, s_bo, nullptr, 0, vreg, cD, cBS, cD, cD);
  slstm_kernel<<<cB, 256, slstmLds, stream>>>(xm, zgb, kreg, vreg, s_Ri, s_Rf, s_Rz, s_Ro, qreg);
  hnorm_add_kernel<<<cBS * cH, 64, 0, stream>>>(qreg, s_ng, s_nb, x);

  // ---- FFN ----
  ln_kernel<<<cBS, 256, 0, stream>>>(x, f_ln_g, f_ln_b, xn, 1e-5f);
  gemm_kernel<false, false><<<gN352, 256, 0, stream>>>(xn, cD, f_Wup,       2 * cFF, f_bup,       nullptr, 0, t1, cFF, cBS, cFF, cD);
  gemm_kernel<false, false><<<gN352, 256, 0, stream>>>(xn, cD, f_Wup + cFF, 2 * cFF, f_bup + cFF, nullptr, 0, ab, cFF, cBS, cFF, cD);
  const int nglu = cBS * cFF;
  gelu_mul_kernel<<<(nglu + 255) / 256, 256, 0, stream>>>(t1, ab, t1, nglu);
  gemm_kernel<true, false><<<gN256, 256, 0, stream>>>(t1, cFF, f_Wdown, cD, f_bdown, x, cD, x, cD, cBS, cD, cFF);

  // ---- head ----
  head_kernel<<<cB, 64, 0, stream>>>(x, h_W1, h_b1, h_W2, h_b2, (float*)d_out);
}
